// Inv2d_3040836845947
// MI455X (gfx1250) — matrix operand order repaired and checked by reading, not compiled, not run on hardware
//
#include <hip/hip_runtime.h>

// Inv2d (involution) for MI455X / gfx1250, wave32, WMMA bf16 path.
// B=16, C=256, H=W=64. GEMM1: 65536x256x256, GEMM2: 65536x256x2304 fused
// with the 3x3 unfold-multiply-reduce so Ker (604MB) never hits HBM.
// Round 2: TDM (tensor_load_to_lds) staging of the Wn tile in k_span_inv
// + global_prefetch of the L2-resident span-weight fragments.

#define B_    16
#define C_    256
#define H_    64
#define W_    64
#define HW_   (H_ * W_)      // 4096
#define NPIX_ (B_ * HW_)     // 65536
#define K2_   9
#define CK2_  (C_ * K2_)     // 2304
#define EPS_  1e-5f

typedef __attribute__((ext_vector_type(16))) __bf16 v16bf;
typedef __attribute__((ext_vector_type(8)))  float  v8f;
typedef __attribute__((ext_vector_type(4)))  unsigned int u32x4;
typedef __attribute__((ext_vector_type(8)))  int          i32x8;
typedef __attribute__((ext_vector_type(4)))  int          i32x4;

union FragU { v16bf v; uint4 q[2]; };

__device__ inline unsigned short f2bf(float f) {
  unsigned int u = __float_as_uint(f);
  u += 0x7FFFu + ((u >> 16) & 1u);   // round-to-nearest-even
  return (unsigned short)(u >> 16);
}
__device__ inline float bf2f(unsigned short h) {
  return __uint_as_float(((unsigned int)h) << 16);
}
__device__ inline v8f wmma_bf16(v16bf a, v16bf b, v8f c) {
  // v_wmma_f32_16x16x32_bf16  (8-arg form, codegen-confirmed)
  return __builtin_amdgcn_wmma_f32_16x16x32_bf16(
      false, a, false, b, (short)0, c, false, false);
}

// ---------------------------------------------------------------------------
// k_prep: convert w_reduce / w_span to bf16 pre-swizzled into the WMMA
// B-fragment layout [nt][kt][lane][16], and zero the BN stats accumulators.
// B 32x16 bf16 layout: lane = (n&15) + 16*((klocal>>3)&1), pos = (klocal&7) + 8*(klocal>>4).
// ---------------------------------------------------------------------------
__global__ __launch_bounds__(256) void k_prep(const float* __restrict__ wr,
                                              const float* __restrict__ wspan,
                                              unsigned short* __restrict__ b1,
                                              unsigned short* __restrict__ b2,
                                              float* __restrict__ stats) {
  int g = blockIdx.x * 256 + threadIdx.x;
  if (g < 512) stats[g] = 0.f;
  if (g < C_ * C_) {
    int o = g >> 8, c = g & 255;
    int nt = o >> 4, kt = c >> 5, kl = c & 31;
    int lane = (o & 15) + 16 * ((kl >> 3) & 1);
    int pos  = (kl & 7) + 8 * (kl >> 4);
    b1[((nt * 8 + kt) * 32 + lane) * 16 + pos] = f2bf(wr[g]);
  }
  if (g < CK2_ * C_) {
    int o = g >> 8, c = g & 255;
    int nt = o >> 4, kt = c >> 5, kl = c & 31;
    int lane = (o & 15) + 16 * ((kl >> 3) & 1);
    int pos  = (kl & 7) + 8 * (kl >> 4);
    b2[((nt * 8 + kt) * 32 + lane) * 16 + pos] = f2bf(wspan[g]);
  }
}

// ---------------------------------------------------------------------------
// k_reduce: GEMM1 (X * w_reduce^T + b_reduce) -> Wr (bf16) + BN partial stats.
// Block = 128 pixels x 256 out-channels, K=256. 8 waves: 4 (M) x 2 (N),
// each wave owns 2 M-tiles x 8 N-tiles (16 accumulators, 128 WMMAs).
// A staged to LDS pre-swizzled into A-fragment layout:
//   A 16x32 bf16: lane=(m&15)+16*((kl>>3)&1), pos=(kl&7)+8*(kl>>4).
// ---------------------------------------------------------------------------
__global__ __launch_bounds__(256) void k_reduce(const float* __restrict__ X,
                                                const unsigned short* __restrict__ B1,
                                                const float* __restrict__ b_reduce,
                                                unsigned short* __restrict__ Wr,
                                                float* __restrict__ stats) {
  __shared__ unsigned short Als[8 * 8 * 32 * 16]; // 64 KB: 8 mt x 8 kt frags
  __shared__ float sred[512];
  const int tid  = threadIdx.x;
  const int pix0 = blockIdx.x * 128;
  const int b    = pix0 >> 12;           // pixels per batch image = 4096
  const int off  = pix0 & (HW_ - 1);

  sred[tid] = 0.f;
  sred[tid + 256] = 0.f;

  // ---- stage A tile (128 pixels x 256 channels) f32 -> bf16 swizzled LDS
  const float* Xb = X + (size_t)b * C_ * HW_ + off;
  #pragma unroll
  for (int i = 0; i < 32; ++i) {
    int e  = i * 256 + tid;              // 8192 float4 loads total
    int k  = e >> 5;                     // channel
    int m4 = (e & 31) * 4;               // pixel
    const float4 f = *(const float4*)(Xb + (size_t)k * HW_ + m4);
    int kt = k >> 5, kl = k & 31;
    int lh  = (kl >> 3) & 1;
    int pos = (kl & 7) + 8 * (kl >> 4);
    float vals[4] = {f.x, f.y, f.z, f.w};
    #pragma unroll
    for (int j = 0; j < 4; ++j) {
      int m = m4 + j;
      int mt = m >> 4;
      int lane = (m & 15) + 16 * lh;
      Als[((mt * 8 + kt) * 32 + lane) * 16 + pos] = f2bf(vals[j]);
    }
  }
  __syncthreads();

  const int lane = tid & 31;
  const int wid  = tid >> 5;
  const int wm   = wid & 3;              // M-tiles {2wm, 2wm+1}
  const int wn   = wid >> 2;             // N-tiles wn*8 .. wn*8+7

  v8f acc[2][8];
  #pragma unroll
  for (int a = 0; a < 2; ++a)
    #pragma unroll
    for (int j = 0; j < 8; ++j) acc[a][j] = v8f{};

  const uint4* Aq = (const uint4*)Als;
  const uint4* Bq = (const uint4*)B1;    // L2-resident, fragment layout
  for (int kt = 0; kt < 8; ++kt) {
    FragU a0, a1;
    a0.q[0] = Aq[(((2 * wm) * 8 + kt) * 32 + lane) * 2 + 0];
    a0.q[1] = Aq[(((2 * wm) * 8 + kt) * 32 + lane) * 2 + 1];
    a1.q[0] = Aq[(((2 * wm + 1) * 8 + kt) * 32 + lane) * 2 + 0];
    a1.q[1] = Aq[(((2 * wm + 1) * 8 + kt) * 32 + lane) * 2 + 1];
    #pragma unroll
    for (int j = 0; j < 8; ++j) {
      int nt = wn * 8 + j;
      FragU bf;
      bf.q[0] = Bq[((nt * 8 + kt) * 32 + lane) * 2 + 0];
      bf.q[1] = Bq[((nt * 8 + kt) * 32 + lane) * 2 + 1];
      acc[0][j] = wmma_bf16(a0.v, bf.v, acc[0][j]);
      acc[1][j] = wmma_bf16(a1.v, bf.v, acc[1][j]);
    }
  }

  // ---- epilogue: + bias, stats, bf16 store (lane owns one column o,
  // 8 consecutive pixels -> one 16B store)
  const int col     = lane & 15;
  const int rowhalf = (lane >> 4) * 8;
  #pragma unroll
  for (int mi = 0; mi < 2; ++mi) {
    int mt = 2 * wm + mi;
    int lp = mt * 16 + rowhalf;
    #pragma unroll
    for (int j = 0; j < 8; ++j) {
      int o = (wn * 8 + j) * 16 + col;
      float br = b_reduce[o];
      float s = 0.f, s2 = 0.f;
      unsigned int pk[4];
      #pragma unroll
      for (int r2 = 0; r2 < 4; ++r2) {
        float v0 = acc[mi][j][2 * r2] + br;
        float v1 = acc[mi][j][2 * r2 + 1] + br;
        s += v0 + v1;
        s2 += v0 * v0 + v1 * v1;
        pk[r2] = (unsigned int)f2bf(v0) | ((unsigned int)f2bf(v1) << 16);
      }
      uint4 q; q.x = pk[0]; q.y = pk[1]; q.z = pk[2]; q.w = pk[3];
      *(uint4*)(Wr + ((size_t)b * C_ + o) * HW_ + off + lp) = q;
      atomicAdd(&sred[o], s);
      atomicAdd(&sred[256 + o], s2);
    }
  }
  __syncthreads();
  atomicAdd(&stats[tid], sred[tid]);
  atomicAdd(&stats[256 + tid], sred[256 + tid]);
}

// ---------------------------------------------------------------------------
// k_bnstats: fold batch stats into per-channel scale/bias.
// ---------------------------------------------------------------------------
__global__ __launch_bounds__(256) void k_bnstats(const float* __restrict__ stats,
                                                 const float* __restrict__ gamma,
                                                 const float* __restrict__ beta,
                                                 float* __restrict__ sb) {
  int c = threadIdx.x;
  float inv  = 1.f / (float)NPIX_;
  float mean = stats[c] * inv;
  float var  = stats[256 + c] * inv - mean * mean;
  float rs   = rsqrtf(var + EPS_);
  float sc   = gamma[c] * rs;
  sb[c]       = sc;
  sb[256 + c] = beta[c] - mean * sc;
}

// ---------------------------------------------------------------------------
// k_span_inv: GEMM2 (Wn * w_span^T + b_span) fused with the involution apply.
// Block = one (b,h) row of 64 pixels. The raw bf16 Wr tile (64 px x 256 ch,
// row stride 4096 elems) is DMA'd into LDS by the Tensor Data Mover while the
// wave zeroes the output accumulator; affine+ReLU+fragment-swizzle then runs
// LDS->LDS. 144 N-tiles of 16 span channels; lane's column = span channel
// s -> (c=s/9, k2=s%9); Ker contracted against X patches into LDS f32 accum.
// ---------------------------------------------------------------------------
__global__ __launch_bounds__(256) void k_span_inv(
    const unsigned short* __restrict__ Wr,
    const unsigned short* __restrict__ B2,
    const float* __restrict__ sb,
    const float* __restrict__ b_span,
    const float* __restrict__ X,
    float* __restrict__ out) {
  __shared__ unsigned short Raw[C_ * W_];         // 32 KB raw Wr tile [k][m]
  __shared__ unsigned short Als[4 * 8 * 32 * 16]; // 32 KB swizzled A frags
  __shared__ float outs[C_ * W_];                 // 64 KB accumulator
  const int tid = threadIdx.x;
  const int row = blockIdx.x;    // 0..1023
  const int b   = row >> 6;
  const int h   = row & 63;
  const unsigned short* Wrb = Wr + (size_t)b * C_ * HW_ + h * W_;

#if __has_builtin(__builtin_amdgcn_tensor_load_to_lds)
  // ---- Tensor Data Mover: 2D tile, data_size=2B, tile 64 x 256,
  //      tensor_dim0_stride = 4096 elems -> contiguous [k][m] rows in LDS.
  if (tid < 32) {   // one wave issues the DMA (EXEC is ignored by TDM)
    unsigned long long ga = (unsigned long long)(uintptr_t)Wrb;
    unsigned int lds_off  = (unsigned int)(uintptr_t)Raw; // flat LDS addr[31:0]
    u32x4 g0;
    g0[0] = 1u;                                   // count=1, user descriptor
    g0[1] = lds_off;                              // lds_addr
    g0[2] = (unsigned int)(ga & 0xFFFFFFFFu);     // global_addr[31:0]
    g0[3] = (unsigned int)((ga >> 32) & 0x1FFFFFFu) | (2u << 30); // [56:32]|type=2
    i32x8 g1;
    g1[0] = (int)(1u << 16);                      // data_size=1 (2 bytes)
    g1[1] = (int)((4096u & 0xFFFFu) << 16);       // tensor_dim0[15:0] @ 63:48
    g1[2] = (int)((256u & 0xFFFFu) << 16);        // dim0 hi=0 | tensor_dim1 lo
    g1[3] = (int)(64u << 16);                     // dim1 hi=0 | tile_dim0=64
    g1[4] = 256;                                  // tile_dim1=256, tile_dim2=0
    g1[5] = 4096;                                 // tensor_dim0_stride[31:0]
    g1[6] = 0;                                    // stride hi | dim1_stride lo
    g1[7] = 0;
    i32x4 g2 = {0, 0, 0, 0};
    i32x4 g3 = {0, 0, 0, 0};
#if __clang_major__ >= 23
    i32x8 g4 = {0, 0, 0, 0, 0, 0, 0, 0};
    __builtin_amdgcn_tensor_load_to_lds(g0, g1, g2, g3, g4, 0);
#else
    __builtin_amdgcn_tensor_load_to_lds(g0, g1, g2, g3, 0);
#endif
  }
#endif

  // zero accumulator while the TDM transfer is in flight
  #pragma unroll
  for (int i = 0; i < 64; ++i) outs[i * 256 + tid] = 0.f;

#if __has_builtin(__builtin_amdgcn_tensor_load_to_lds)
  if (tid < 32) __builtin_amdgcn_s_wait_tensorcnt(0);
#else
  // fallback: plain wave-parallel copy global -> Raw
  #pragma unroll
  for (int i = 0; i < 8; ++i) {
    int e  = i * 256 + tid;
    int k  = e >> 3;
    int m8 = (e & 7) * 8;
    *(uint4*)(Raw + k * W_ + m8) = *(const uint4*)(Wrb + (size_t)k * HW_ + m8);
  }
#endif
  __syncthreads();

  // ---- affine + ReLU + swizzle into A-fragment layout (LDS -> LDS)
  #pragma unroll
  for (int i = 0; i < 8; ++i) {
    int e  = i * 256 + tid;
    int k  = e >> 3;
    int m8 = (e & 7) * 8;
    uint4 q = *(const uint4*)(Raw + k * W_ + m8);
    float sc = sb[k], bi = sb[256 + k];
    unsigned int wsv[4] = {q.x, q.y, q.z, q.w};
    int kt = k >> 5, kl = k & 31;
    int lh  = (kl >> 3) & 1;
    int pos = (kl & 7) + 8 * (kl >> 4);
    #pragma unroll
    for (int j = 0; j < 8; ++j) {
      unsigned short hv = (unsigned short)((wsv[j >> 1] >> ((j & 1) * 16)) & 0xFFFFu);
      float v = bf2f(hv) * sc + bi;
      v = v > 0.f ? v : 0.f;                       // ReLU
      int m = m8 + j;
      int mt = m >> 4;
      int lane = (m & 15) + 16 * lh;
      Als[((mt * 8 + kt) * 32 + lane) * 16 + pos] = f2bf(v);
    }
  }
  __syncthreads();

  const int lane = tid & 31;
  const int wid  = tid >> 5;
  const int mt   = wid & 3;      // wave's M-tile (16 pixels)
  const int nt0  = wid >> 2;     // N-tile parity (stride 2 over 144 tiles)
  const uint4* Aq = (const uint4*)Als;
  const uint4* Bq = (const uint4*)B2;

  // hoist the wave's 8 A fragments (reused for all 72 N-tiles)
  FragU afr[8];
  #pragma unroll
  for (int kt = 0; kt < 8; ++kt) {
    afr[kt].q[0] = Aq[((mt * 8 + kt) * 32 + lane) * 2 + 0];
    afr[kt].q[1] = Aq[((mt * 8 + kt) * 32 + lane) * 2 + 1];
  }

  const int col   = lane & 15;
  const int wbase = mt * 16 + (lane >> 4) * 8;
  for (int nt = nt0; nt < 144; nt += 2) {
    if (nt + 2 < 144) {  // prefetch next N-tile's fragments (L2-resident)
      __builtin_prefetch(Bq + ((size_t)((nt + 2) * 8) * 32 + lane) * 2, 0, 0);
    }
    v8f acc = v8f{};
    #pragma unroll
    for (int kt = 0; kt < 8; ++kt) {
      FragU bf;
      bf.q[0] = Bq[((size_t)(nt * 8 + kt) * 32 + lane) * 2 + 0];
      bf.q[1] = Bq[((size_t)(nt * 8 + kt) * 32 + lane) * 2 + 1];
      acc = wmma_bf16(afr[kt].v, bf.v, acc);
    }
    // fused involution epilogue: lane's column = span channel s
    int s  = nt * 16 + col;
    int c  = s / 9;
    int k2 = s - c * 9;
    int di = k2 / 3 - 1;
    int dj = (k2 - (k2 / 3) * 3) - 1;
    float bias = b_span[s];
    int hh = h + di;
    bool hok = (hh >= 0) && (hh < H_);
    const float* Xc = X + ((size_t)b * C_ + c) * HW_ + hh * W_;
    #pragma unroll
    for (int r = 0; r < 8; ++r) {
      int wp  = wbase + r;
      int wwx = wp + dj;
      float xv = 0.f;
      if (hok && wwx >= 0 && wwx < W_) xv = Xc[wwx];
      float kerv = acc[r] + bias;
      atomicAdd(&outs[c * W_ + wp], kerv * xv);   // ds_add_f32
    }
  }
  __syncthreads();

  // ---- writeout: thread tid owns channel tid's 64 pixels (coalesced f32x4)
  float* ob = out + ((size_t)b * C_ + tid) * HW_ + h * W_;
  const float* sl = &outs[tid * W_];
  #pragma unroll
  for (int i = 0; i < 16; ++i)
    ((float4*)ob)[i] = ((const float4*)sl)[i];
}

// ---------------------------------------------------------------------------
extern "C" void kernel_launch(void* const* d_in, const int* in_sizes, int n_in,
                              void* d_out, int out_size, void* d_ws, size_t ws_size,
                              hipStream_t stream) {
  const float* X        = (const float*)d_in[0];
  const float* w_reduce = (const float*)d_in[1];
  const float* b_reduce = (const float*)d_in[2];
  const float* gamma    = (const float*)d_in[3];
  const float* beta     = (const float*)d_in[4];
  const float* w_span   = (const float*)d_in[5];
  const float* b_span   = (const float*)d_in[6];

  char* ws = (char*)d_ws;
  float*          stats = (float*)ws;                          //   2 KB
  float*          sb    = (float*)(ws + 2048);                 //   2 KB
  unsigned short* B1    = (unsigned short*)(ws + 4096);        // 128 KB
  unsigned short* B2    = (unsigned short*)(ws + 4096 + 131072);        // 1.125 MB
  unsigned short* Wr    = (unsigned short*)(ws + 4096 + 131072 + 1179648); // 32 MB

  k_prep    <<<2304, 256, 0, stream>>>(w_reduce, w_span, B1, B2, stats);
  k_reduce  <<< 512, 256, 0, stream>>>(X, B1, b_reduce, Wr, stats);
  k_bnstats <<<   1, 256, 0, stream>>>(stats, gamma, beta, sb);
  k_span_inv<<<1024, 256, 0, stream>>>(Wr, B2, sb, b_span, X, (float*)d_out);
}
